// MLAFormer_18786186953481
// MI455X (gfx1250) — compile-verified, matrix operand order checked by reading, not run on hardware
//
#include <hip/hip_runtime.h>
#include <hip/hip_bf16.h>

// ---------------- types ----------------
typedef __bf16 bf16;
typedef __attribute__((ext_vector_type(8)))  __bf16 v8bf;
typedef __attribute__((ext_vector_type(16))) __bf16 v16bf;
typedef __attribute__((ext_vector_type(8)))  float  v8f;

// Problem constants
#define BB      2
#define LL      1024
#define DIMC    512
#define HEADS   8
#define DH      64
#define CATC    1536   // 3*DIM
#define LFORE   4096
#define LPOST   256
#define FORED   256
#define POSTD   1024
#define KVC     1024   // 2*DIM

__device__ __forceinline__ bf16 f2bf(float f) { return (bf16)f; }  // native v_cvt

// Build a 16-element bf16 fragment from two contiguous 8-element groups.
__device__ __forceinline__ v16bf ldfrag(const bf16* p, int o0, int o1) {
    v8bf a = *reinterpret_cast<const v8bf*>(p + o0);
    v8bf b = *reinterpret_cast<const v8bf*>(p + o1);
    return __builtin_shufflevector(a, b, 0,1,2,3,4,5,6,7,8,9,10,11,12,13,14,15);
}

__device__ __forceinline__ v8f wmma_bf16(v16bf a, v16bf b, v8f c) {
    return __builtin_amdgcn_wmma_f32_16x16x32_bf16(false, a, false, b, (short)0, c,
                                                   false, false);
}

// ---------------- f32 -> bf16 conversion (optionally strided dest) ----------
__global__ __launch_bounds__(256) void cvt_bf16_kernel(const float* __restrict__ src,
                                                       bf16* __restrict__ dst,
                                                       int ncols, int ldo, int ocol,
                                                       int total) {
    int i = blockIdx.x * 256 + threadIdx.x;
    if (i >= total) return;
    int r = i / ncols, c = i - r * ncols;
    dst[(size_t)r * ldo + ocol + c] = f2bf(src[i]);
}

// ---------------- generic WMMA GEMM --------------------------------------
// OUT = (A_bf16[M,K] @ W_f32[K,N] + bias) * alpha   (bias must be non-null)
// Block: 256 threads = 8 waves; wave tile 16x64; block tile 64(M) x 128(N).
__global__ __launch_bounds__(256) void gemm_wmma_kernel(
    const bf16* __restrict__ A, int lda,
    const float* __restrict__ W, const float* __restrict__ bias,
    float* __restrict__ outF, bf16* __restrict__ outB,
    int N, int K, int ldo, int ocol, float alpha) {

    const int lane = threadIdx.x & 31;
    const int wave = threadIdx.x >> 5;
    const int half = lane >> 4;
    const int l16  = lane & 15;
    const int wm = wave & 3;          // 4 waves stacked in M
    const int wn = wave >> 2;         // 2 waves in N
    const int m0   = blockIdx.x * 64 + wm * 16;
    const int nblk = blockIdx.y * 128;
    const int nw   = nblk + wn * 64;

    __shared__ __align__(16) bf16 WT[128][32];   // transposed W tile (bf16)

    v8f acc[4] = {{}, {}, {}, {}};
    const bf16* arow = A + (size_t)(m0 + l16) * lda;

    for (int k0 = 0; k0 < K; k0 += 32) {
        __syncthreads();
        // stage W[k0..k0+31, nblk..nblk+127] transposed into LDS as bf16
        #pragma unroll
        for (int i = 0; i < 4; ++i) {
            int idx4 = i * 256 + threadIdx.x;   // 0..1023 float4s
            int kr = idx4 >> 5;                 // 0..31
            int n4 = (idx4 & 31) * 4;           // 0..124
            const float4 wv =
                *reinterpret_cast<const float4*>(&W[(size_t)(k0 + kr) * N + nblk + n4]);
            WT[n4 + 0][kr] = f2bf(wv.x);
            WT[n4 + 1][kr] = f2bf(wv.y);
            WT[n4 + 2][kr] = f2bf(wv.z);
            WT[n4 + 3][kr] = f2bf(wv.w);
        }
        __syncthreads();

        // A fragment: lane<16 holds row l16, K = k0+{0..7,16..23}; lane>=16 K=+8
        v16bf af = ldfrag(arow + k0, half * 8, 16 + half * 8);
        v16bf bfr[4];
        #pragma unroll
        for (int nt = 0; nt < 4; ++nt)
            bfr[nt] = ldfrag(&WT[wn * 64 + nt * 16 + l16][0], half * 8, 16 + half * 8);
        #pragma unroll
        for (int nt = 0; nt < 4; ++nt)
            acc[nt] = wmma_bf16(af, bfr[nt], acc[nt]);
    }

    // epilogue: hoist bias loads; uniform output-format branch outside the loops
    float bl[4];
    #pragma unroll
    for (int nt = 0; nt < 4; ++nt) bl[nt] = bias[nw + nt * 16 + l16];

    const int rbase = m0 + 8 * half;                 // C layout: M=r (lanes<16), r+8
    if (outB) {
        bf16* ob = outB + (size_t)rbase * ldo + ocol + nw + l16;
        #pragma unroll
        for (int nt = 0; nt < 4; ++nt)
            #pragma unroll
            for (int r = 0; r < 8; ++r)
                ob[(size_t)r * ldo + nt * 16] = f2bf((acc[nt][r] + bl[nt]) * alpha);
    } else {
        float* of = outF + (size_t)rbase * ldo + nw + l16;
        #pragma unroll
        for (int nt = 0; nt < 4; ++nt)
            #pragma unroll
            for (int r = 0; r < 8; ++r)
                __builtin_nontemporal_store((acc[nt][r] + bl[nt]) * alpha,
                                            &of[(size_t)r * ldo + nt * 16]);
    }
}

// ---------------- flash cross-attention ----------------------------------
// grid = (L/128, HEADS, B); block = 256 (8 waves, wave owns 16 q rows).
// q: bf16 [B*L, DIM] pre-scaled; kv: bf16 [B*Lc, 1024] (k cols 0..511, v 512..1023)
// writes bf16 into cat[B*L, 1536] at column base ocol.
__global__ __launch_bounds__(256) void flash_attn_kernel(
    const bf16* __restrict__ q, const bf16* __restrict__ kv,
    bf16* __restrict__ cat, int Lc, int ocol) {

    const int lane = threadIdx.x & 31;
    const int wave = threadIdx.x >> 5;
    const int half = lane >> 4;
    const int l16  = lane & 15;
    const int qb = blockIdx.x, h = blockIdx.y, b = blockIdx.z;

    __shared__ __align__(16) bf16 Ksh[32][64];     // K chunk, row-major (async-filled)
    __shared__ __align__(16) bf16 VT[64][32];      // V^T tile, shared by block
    __shared__ __align__(16) bf16 P[8][16][32];    // per-wave P staging

    // LDS byte offset of Ksh for the async-copy instruction
    unsigned kshBase =
        (unsigned)(unsigned long long)(__attribute__((address_space(3))) char*)Ksh;

    const int qrow0 = qb * 128 + wave * 16;
    const bf16* qrow = q + (size_t)(b * LL + qrow0 + l16) * DIMC + h * DH;
    // Q A-fragments for the two 32-wide K-steps over D=64 (resident whole kernel)
    v16bf aq0 = ldfrag(qrow,      half * 8, 16 + half * 8);
    v16bf aq1 = ldfrag(qrow, 32 + half * 8, 48 + half * 8);

    const bf16* kbase = kv + (size_t)b * Lc * KVC + h * DH;        // k part
    const bf16* vbase = kbase + DIMC;                               // v part

    float m[8], l[8];
    v8f o[4] = {{}, {}, {}, {}};
    #pragma unroll
    for (int r = 0; r < 8; ++r) { m[r] = -1e30f; l[r] = 0.f; }

    for (int c0 = 0; c0 < Lc; c0 += 32) {
        __syncthreads();   // previous iteration's LDS reads done

        // ---- async-stage K chunk (32 rows x 64 d, row-major) into LDS ----
        {
            int t = threadIdx.x;                      // 256 threads x 16B = 4KB
            const bf16* gk = kbase + (size_t)(c0 + (t >> 3)) * KVC + (t & 7) * 8;
            unsigned ldst = kshBase + (unsigned)t * 16u;
            asm volatile("global_load_async_to_lds_b128 %0, %1, off"
                         :: "v"(ldst), "v"((unsigned long long)gk)
                         : "memory");
        }

        // ---- stage V^T (32 kv rows x 64 d) into LDS, transposed ----
        {
            int kvr = threadIdx.x >> 3;               // 0..31
            int d0  = (threadIdx.x & 7) * 8;          // 0..56
            v8bf vv = *reinterpret_cast<const v8bf*>(
                vbase + (size_t)(c0 + kvr) * KVC + d0);
            #pragma unroll
            for (int i = 0; i < 8; ++i) VT[d0 + i][kvr] = vv[i];
        }
        if (c0 + 32 < Lc)
            __builtin_prefetch(vbase + (size_t)(c0 + 32) * KVC, 0, 0);

        asm volatile("s_wait_asynccnt 0" ::: "memory");
        __syncthreads();

        // ---- S = Q K^T for 32 kv columns (two 16x16 C tiles) ----
        v16bf kf[4];
        #pragma unroll
        for (int ct = 0; ct < 2; ++ct)
            #pragma unroll
            for (int ks = 0; ks < 2; ++ks)
                kf[ct * 2 + ks] =
                    ldfrag(&Ksh[ct * 16 + l16][ks * 32], half * 8, 16 + half * 8);
        v8f s[2];
        #pragma unroll
        for (int ct = 0; ct < 2; ++ct) {
            v8f a = {};
            a = wmma_bf16(aq0, kf[ct * 2 + 0], a);
            a = wmma_bf16(aq1, kf[ct * 2 + 1], a);
            s[ct] = a;
        }

        // ---- online softmax over the 32 new columns ----
        float alpha[8];
        #pragma unroll
        for (int r = 0; r < 8; ++r) {
            float rm = fmaxf(s[0][r], s[1][r]);
            #pragma unroll
            for (int mk = 1; mk < 16; mk <<= 1)
                rm = fmaxf(rm, __shfl_xor(rm, mk, 32));
            float mn = fmaxf(m[r], rm);
            alpha[r] = __expf(m[r] - mn);
            m[r] = mn;
        }
        #pragma unroll
        for (int r = 0; r < 8; ++r) {
            float p0 = __expf(s[0][r] - m[r]);
            float p1 = __expf(s[1][r] - m[r]);
            int row = r + 8 * half;                    // C layout row
            P[wave][row][l16]      = f2bf(p0);
            P[wave][row][16 + l16] = f2bf(p1);
            float rs = p0 + p1;
            #pragma unroll
            for (int mk = 1; mk < 16; mk <<= 1)
                rs += __shfl_xor(rs, mk, 32);
            l[r] = l[r] * alpha[r] + rs;
        }
        #pragma unroll
        for (int nt = 0; nt < 4; ++nt)
            #pragma unroll
            for (int r = 0; r < 8; ++r)
                o[nt][r] *= alpha[r];

        asm volatile("s_wait_dscnt 0" ::: "memory");   // P stores visible wave-wide

        // ---- O += P x V (P relayout C->A via LDS; V^T rows are contiguous) ----
        const bf16* prow = &P[wave][l16][0];
        v16bf pa = ldfrag(prow, half * 8, 16 + half * 8);
        v16bf bv[4];
        #pragma unroll
        for (int nt = 0; nt < 4; ++nt)
            bv[nt] = ldfrag(&VT[nt * 16 + l16][0], half * 8, 16 + half * 8);
        #pragma unroll
        for (int nt = 0; nt < 4; ++nt)
            o[nt] = wmma_bf16(pa, bv[nt], o[nt]);
    }

    // ---- normalize (v_rcp_f32 + mul, not 32 divides) and write bf16 ----
    float inv[8];
    #pragma unroll
    for (int r = 0; r < 8; ++r) inv[r] = __builtin_amdgcn_rcpf(l[r]);

    bf16* ob = cat + (size_t)(b * LL + qrow0 + 8 * half) * CATC + ocol + h * DH + l16;
    #pragma unroll
    for (int nt = 0; nt < 4; ++nt)
        #pragma unroll
        for (int r = 0; r < 8; ++r)
            ob[(size_t)r * CATC + nt * 16] = f2bf(o[nt][r] * inv[r]);
}

// ---------------- launch ---------------------------------------------------
extern "C" void kernel_launch(void* const* d_in, const int* in_sizes, int n_in,
                              void* d_out, int out_size, void* d_ws, size_t ws_size,
                              hipStream_t stream) {
    (void)in_sizes; (void)n_in; (void)out_size; (void)ws_size;
    const float* x      = (const float*)d_in[0];
    const float* fore_x = (const float*)d_in[1];
    const float* post_x = (const float*)d_in[2];
    const float* wq     = (const float*)d_in[3];
    const float* bq     = (const float*)d_in[4];
    const float* wkv1   = (const float*)d_in[5];
    const float* bkv1   = (const float*)d_in[6];
    const float* wkv2   = (const float*)d_in[7];
    const float* bkv2   = (const float*)d_in[8];
    const float* wproj  = (const float*)d_in[9];
    const float* bproj  = (const float*)d_in[10];
    float* out = (float*)d_out;

    // workspace layout (bf16 buffers, all 16B-aligned offsets)
    char* ws = (char*)d_ws;
    size_t off = 0;
    bf16* catb  = (bf16*)(ws + off); off += (size_t)BB * LL * CATC * 2;      // 6.0 MB
    bf16* foreb = (bf16*)(ws + off); off += (size_t)BB * LFORE * FORED * 2;  // 4.0 MB
    bf16* postb = (bf16*)(ws + off); off += (size_t)BB * LPOST * POSTD * 2;  // 1.0 MB
    bf16* qb8   = (bf16*)(ws + off); off += (size_t)BB * LL * DIMC * 2;      // 2.0 MB
    bf16* kv1b  = (bf16*)(ws + off); off += (size_t)BB * LFORE * KVC * 2;    // 16  MB
    bf16* kv2b  = (bf16*)(ws + off); off += (size_t)BB * LPOST * KVC * 2;    // 1.0 MB

    const float scale = 0.125f;  // D^-0.5 = 64^-0.5

    // 1) f32 -> bf16 conversions
    {
        int t = BB * LL * DIMC;                       // x -> cat cols [0,512)
        cvt_bf16_kernel<<<t / 256, 256, 0, stream>>>(x, catb, DIMC, CATC, 0, t);
        t = BB * LFORE * FORED;
        cvt_bf16_kernel<<<t / 256, 256, 0, stream>>>(fore_x, foreb, FORED, FORED, 0, t);
        t = BB * LPOST * POSTD;
        cvt_bf16_kernel<<<t / 256, 256, 0, stream>>>(post_x, postb, POSTD, POSTD, 0, t);
    }

    // 2) projections (WMMA GEMMs)
    // q = (x@wq + bq) * scale  -> bf16 [2048, 512]
    gemm_wmma_kernel<<<dim3((BB * LL) / 64, DIMC / 128), 256, 0, stream>>>(
        catb, CATC, wq, bq, nullptr, qb8, DIMC, DIMC, DIMC, 0, scale);
    // kv1 = fore_x@wkv1 + bkv1 -> bf16 [8192, 1024]
    gemm_wmma_kernel<<<dim3((BB * LFORE) / 64, KVC / 128), 256, 0, stream>>>(
        foreb, FORED, wkv1, bkv1, nullptr, kv1b, KVC, FORED, KVC, 0, 1.0f);
    // kv2 = post_x@wkv2 + bkv2 -> bf16 [512, 1024]
    gemm_wmma_kernel<<<dim3((BB * LPOST) / 64, KVC / 128), 256, 0, stream>>>(
        postb, POSTD, wkv2, bkv2, nullptr, kv2b, KVC, POSTD, KVC, 0, 1.0f);

    // 3) cross-attentions -> cat cols [512,1024) and [1024,1536)
    dim3 agrid(LL / 128, HEADS, BB);
    flash_attn_kernel<<<agrid, 256, 0, stream>>>(qb8, kv1b, catb, LFORE, DIMC);
    flash_attn_kernel<<<agrid, 256, 0, stream>>>(qb8, kv2b, catb, LPOST, 2 * DIMC);

    // 4) out = cat @ wproj + bproj  (f32 output)
    gemm_wmma_kernel<<<dim3((BB * LL) / 64, DIMC / 128), 256, 0, stream>>>(
        catb, CATC, wproj, bproj, out, nullptr, DIMC, CATC, DIMC, 0, 1.0f);
}
